// MolGAT_36644660969537
// MI455X (gfx1250) — matrix operand order repaired and checked by reading, not compiled, not run on hardware
//
#include <hip/hip_runtime.h>
#include <math.h>

// MolGAT for MI455X (gfx1250, wave32).
// - WMMA f16->f32 GEMMs for all node transforms (v_wmma_f32_16x16x32_f16).
// - Block-2 GEMM stages A-tiles via global_load_async_to_lds_b128 (async DMA,
//   ASYNCcnt) since h1 rows are 512B = 32 aligned 16B chunks.
// - 3-pass segment softmax with L2-resident float atomics (agg = 51MB << 192MB L2).
// - Edge embedding (13->128) recomputed inline to avoid 256MB of traffic.
// - Fused ELU+residual+LayerNorm per node; pooling fused into block-2 post.

#define D 128
#define HNUM 4
#define NODE_DIM 38
#define EDGE_DIM 13
#define DEC_HID 64
#define NEG_SLOPE 0.2f
#define LN_EPS 1e-5f

#define AS1 __attribute__((address_space(1)))
#define AS3 __attribute__((address_space(3)))

typedef __attribute__((ext_vector_type(16))) _Float16 v16h;
typedef __attribute__((ext_vector_type(8)))  float    v8f;
typedef __attribute__((ext_vector_type(4)))  int      v4i;

// order-preserving float <-> uint key for atomicMax on floats
__device__ __forceinline__ unsigned f2key(float f) {
  unsigned u = __float_as_uint(f);
  return (u & 0x80000000u) ? ~u : (u | 0x80000000u);
}
__device__ __forceinline__ float key2f(unsigned k) {
  return (k & 0x80000000u) ? __uint_as_float(k & 0x7fffffffu)
                           : __uint_as_float(~k);
}

// ---------------------------------------------------------------------------
// Block-1 GEMM: Y[n,128] = X[n,38] @ W[38,128] + bias. K padded to 64.
// 256 threads = 8 waves; 64 rows/WG; wave w owns 16-col tile w, reuses its
// B fragments across 4 row tiles. Layouts per CDNA5 ISA.
// ---------------------------------------------------------------------------
__global__ __launch_bounds__(256) void gemm_k38_wmma_kernel(
    const float* __restrict__ X, const float* __restrict__ W,
    const float* __restrict__ bias, float* __restrict__ Y, int n_rows) {
  constexpr int KP = 64, KPAD = KP + 8;
  __shared__ _Float16 A_lds[64][KPAD];    // [row][k]
  __shared__ _Float16 Wt_lds[128][KPAD];  // [col][k]
  const int tid = threadIdx.x;
  const int row0 = blockIdx.x * 64;
  const bool full = (row0 + 64 <= n_rows);  // uniform

  {  // stage A as f16, zero-pad k>=38 (and rows >= n_rows on the tail block)
    int r = tid >> 2, q = tid & 3;
    int grow = row0 + r;
    bool rok = full || (grow < n_rows);
#pragma unroll
    for (int k = q; k < KP; k += 4) {
      float v = (rok && k < NODE_DIM) ? X[(size_t)grow * NODE_DIM + k] : 0.f;
      A_lds[r][k] = (_Float16)v;
    }
  }
  {  // stage W transposed
    int col = tid & 127, k0 = tid >> 7;
#pragma unroll
    for (int k = k0; k < KP; k += 2) {
      float v = (k < NODE_DIM) ? W[(size_t)k * D + col] : 0.f;
      Wt_lds[col][k] = (_Float16)v;
    }
  }
  __syncthreads();

  const int wave = tid >> 5;
  const int lane = tid & 31;
  const int colN = wave * 16 + (lane & 15);
  const int hi = lane >> 4;

  v16h bfrag[2];  // B: lane=col colN; i<8 -> K=s*32+hi*8+i ; i>=8 -> K=s*32+16+hi*8+(i-8)
#pragma unroll
  for (int s = 0; s < 2; ++s) {
    int kb = s * 32 + hi * 8;
#pragma unroll
    for (int i = 0; i < 8; ++i) {
      bfrag[s][i] = Wt_lds[colN][kb + i];
      bfrag[s][i + 8] = Wt_lds[colN][kb + 16 + i];
    }
  }
  const float bc = bias[colN];

#pragma unroll
  for (int rt = 0; rt < 4; ++rt) {
    v8f acc;
#pragma unroll
    for (int r = 0; r < 8; ++r) acc[r] = bc;
    const int arow = rt * 16 + (lane & 15);
#pragma unroll
    for (int s = 0; s < 2; ++s) {
      v16h afrag;  // i<8 -> K=s*32+hi*8+i ; i>=8 -> K=s*32+16+hi*8+(i-8)
      int kb = s * 32 + hi * 8;
#pragma unroll
      for (int i = 0; i < 8; ++i) {
        afrag[i] = A_lds[arow][kb + i];
        afrag[i + 8] = A_lds[arow][kb + 16 + i];
      }
      acc = __builtin_amdgcn_wmma_f32_16x16x32_f16(
          false, afrag, false, bfrag[s], (short)0, acc, false, false);
    }
    const int rbase = row0 + rt * 16 + 8 * hi;  // D: acc[r] -> row rbase + r
    float* yp = Y + (size_t)rbase * D + colN;
    if (full) {
#pragma unroll
      for (int r = 0; r < 8; ++r) yp[(size_t)r * D] = acc[r];
    } else {
#pragma unroll
      for (int r = 0; r < 8; ++r)
        if (rbase + r < n_rows) yp[(size_t)r * D] = acc[r];
    }
  }
}

// ---------------------------------------------------------------------------
// Block-2 GEMM: Y[n,128] = X[n,128] @ W[128,128] + bias.
// A-tiles staged into LDS as raw f32 via async DMA (global_load_async_to_lds,
// ASYNCcnt); f32->f16 conversion happens during fragment build. 32 rows/WG.
// Rows >= n_rows skip the DMA; their (garbage) LDS rows only influence output
// rows that are never stored (WMMA rows are independent).
// ---------------------------------------------------------------------------
__global__ __launch_bounds__(256) void gemm_k128_wmma_async_kernel(
    const float* __restrict__ X, const float* __restrict__ W,
    const float* __restrict__ bias, float* __restrict__ Y, int n_rows) {
  constexpr int K = 128;
  __shared__ __align__(16) float A_lds[32][K + 4];  // row stride 528B (16B mult)
  __shared__ _Float16 Wt_lds[128][K + 8];
  const int tid = threadIdx.x;
  const int row0 = blockIdx.x * 32;
  const bool full = (row0 + 32 <= n_rows);  // uniform

  // async stage A: 32 rows x 32 chunks of 16B; 4 chunks per thread
#pragma unroll
  for (int i = 0; i < 4; ++i) {
    int ch = tid + i * 256;
    int r = ch >> 5, q = ch & 31;
    if (full || (row0 + r < n_rows)) {
      AS1 v4i* src = (AS1 v4i*)(X + (size_t)(row0 + r) * K + q * 4);
      AS3 v4i* dst = (AS3 v4i*)(&A_lds[r][q * 4]);
      __builtin_amdgcn_global_load_async_to_lds_b128(src, dst, 0, 0);
    }
  }
  {  // stage W transposed as f16 (overlaps with the DMA)
    int col = tid & 127, k0 = tid >> 7;
#pragma unroll
    for (int k = k0; k < K; k += 2) Wt_lds[col][k] = (_Float16)W[(size_t)k * D + col];
  }
  asm volatile("s_wait_asynccnt 0x0" ::: "memory");
  __syncthreads();

  const int wave = tid >> 5;
  const int lane = tid & 31;
  const int colN = wave * 16 + (lane & 15);
  const int hi = lane >> 4;

  v16h bfrag[4];
#pragma unroll
  for (int s = 0; s < 4; ++s) {
    int kb = s * 32 + hi * 8;
#pragma unroll
    for (int i = 0; i < 8; ++i) {
      bfrag[s][i] = Wt_lds[colN][kb + i];
      bfrag[s][i + 8] = Wt_lds[colN][kb + 16 + i];
    }
  }
  const float bc = bias[colN];

#pragma unroll
  for (int rt = 0; rt < 2; ++rt) {
    v8f acc;
#pragma unroll
    for (int r = 0; r < 8; ++r) acc[r] = bc;
    const int arow = rt * 16 + (lane & 15);
#pragma unroll
    for (int s = 0; s < 4; ++s) {
      v16h afrag;
      int kb = s * 32 + hi * 8;
#pragma unroll
      for (int i = 0; i < 8; ++i) {
        afrag[i] = (_Float16)A_lds[arow][kb + i];
        afrag[i + 8] = (_Float16)A_lds[arow][kb + 16 + i];
      }
      acc = __builtin_amdgcn_wmma_f32_16x16x32_f16(
          false, afrag, false, bfrag[s], (short)0, acc, false, false);
    }
    const int rbase = row0 + rt * 16 + 8 * hi;
    float* yp = Y + (size_t)rbase * D + colN;
    if (full) {
#pragma unroll
      for (int r = 0; r < 8; ++r) yp[(size_t)r * D] = acc[r];
    } else {
#pragma unroll
      for (int r = 0; r < 8; ++r)
        if (rbase + r < n_rows) yp[(size_t)r * D] = acc[r];
    }
  }
}

// ---------------------------------------------------------------------------
// self-loop attrs: deg[dst]++, loop_attr[dst] += edge_attr  (16 threads/edge)
// ---------------------------------------------------------------------------
__global__ void deg_loopattr_kernel(const int* __restrict__ ei,
                                    const float* __restrict__ eattr,
                                    float* __restrict__ deg,
                                    float* __restrict__ lattr, int E_) {
  int t = blockIdx.x * blockDim.x + threadIdx.x;
  int e = t >> 4, k = t & 15;
  if (e >= E_) return;
  int d = ei[E_ + e];
  if (k < EDGE_DIM)
    atomicAdd(&lattr[(size_t)d * EDGE_DIM + k], eattr[(size_t)e * EDGE_DIM + k]);
  else if (k == EDGE_DIM)
    atomicAdd(&deg[d], 1.f);
}

__global__ void loop_div_kernel(float* __restrict__ lattr,
                                const float* __restrict__ deg, int N_) {
  int t = blockIdx.x * blockDim.x + threadIdx.x;
  if (t >= N_ * EDGE_DIM) return;
  lattr[t] /= fmaxf(deg[t / EDGE_DIM], 1.f);
}

// ---------------------------------------------------------------------------
// pass 1: per-edge logits + running segment max. One wave per edge,
// 4 channels/lane; head h reduced over lanes 8h..8h+7.
// ---------------------------------------------------------------------------
__global__ __launch_bounds__(256) void edge_logits_kernel(
    const int* __restrict__ ei, const float* __restrict__ eattr,
    const float* __restrict__ lattr, const float* __restrict__ xl,
    const float* __restrict__ xr, const float* __restrict__ We,
    const float* __restrict__ att, float* __restrict__ logits,
    unsigned* __restrict__ mkey, int E_, int N_) {
  int e = blockIdx.x * 8 + (threadIdx.x >> 5);
  int lane = threadIdx.x & 31;
  if (e >= E_ + N_) return;
  int s, d;
  const float* ea;
  if (e < E_) { s = ei[e]; d = ei[E_ + e]; ea = eattr + (size_t)e * EDGE_DIM; }
  else        { s = e - E_; d = s;         ea = lattr + (size_t)(e - E_) * EDGE_DIM; }
  float eav[EDGE_DIM];
#pragma unroll
  for (int k = 0; k < EDGE_DIM; ++k) eav[k] = ea[k];
  const int c0 = lane * 4;
  const float4 xlv = *(const float4*)(xl + (size_t)s * D + c0);
  const float4 xrv = *(const float4*)(xr + (size_t)d * D + c0);
  const float xls[4] = {xlv.x, xlv.y, xlv.z, xlv.w};
  const float xrs[4] = {xrv.x, xrv.y, xrv.z, xrv.w};
  float part = 0.f;
#pragma unroll
  for (int j = 0; j < 4; ++j) {
    int c = c0 + j;
    float eec = 0.f;  // edge embedding recomputed inline (13 FMAs)
#pragma unroll
    for (int k = 0; k < EDGE_DIM; ++k) eec = fmaf(eav[k], We[k * D + c], eec);
    float z = xls[j] + xrs[j] + eec;
    z = (z > 0.f) ? z : NEG_SLOPE * z;  // leaky_relu
    part = fmaf(z, att[c], part);
  }
  part += __shfl_xor(part, 1, 32);
  part += __shfl_xor(part, 2, 32);
  part += __shfl_xor(part, 4, 32);
  if ((lane & 7) == 0) {
    int h = lane >> 3;
    logits[(size_t)e * HNUM + h] = part;
    atomicMax(&mkey[(size_t)d * HNUM + h], f2key(part));
  }
}

// pass 2: exp(logit - max) in place + segment sum
__global__ void edge_exp_kernel(const int* __restrict__ ei,
                                float* __restrict__ logits,
                                const unsigned* __restrict__ mkey,
                                float* __restrict__ ssum, int E_, int N_) {
  int t = blockIdx.x * blockDim.x + threadIdx.x;
  if (t >= (E_ + N_) * HNUM) return;
  int e = t >> 2, h = t & 3;
  int d = (e < E_) ? ei[E_ + e] : (e - E_);
  float ex = expf(logits[t] - key2f(mkey[(size_t)d * HNUM + h]));
  logits[t] = ex;
  atomicAdd(&ssum[(size_t)d * HNUM + h], ex);
}

// pass 3: agg[dst] += alpha * xl[src]  (L2-resident scatter-add)
__global__ __launch_bounds__(256) void edge_agg_kernel(
    const int* __restrict__ ei, const float* __restrict__ logits,
    const float* __restrict__ ssum, const float* __restrict__ xl,
    float* __restrict__ agg, int E_, int N_) {
  int e = blockIdx.x * 8 + (threadIdx.x >> 5);
  int lane = threadIdx.x & 31;
  if (e >= E_ + N_) return;
  int s, d;
  if (e < E_) { s = ei[e]; d = ei[E_ + e]; } else { s = e - E_; d = s; }
  int h = lane >> 3;
  float w = logits[(size_t)e * HNUM + h] / ssum[(size_t)d * HNUM + h];
  int c0 = lane * 4;
  const float4 xlv = *(const float4*)(xl + (size_t)s * D + c0);
  const float xls[4] = {xlv.x, xlv.y, xlv.z, xlv.w};
#pragma unroll
  for (int j = 0; j < 4; ++j)
    atomicAdd(&agg[(size_t)d * D + c0 + j], xls[j] * w);
}

// ---------------------------------------------------------------------------
// fused ELU + residual + LayerNorm (wave per node); optionally fuses the
// global-mean-pool accumulation (block 2).
// ---------------------------------------------------------------------------
__global__ __launch_bounds__(256) void node_post_kernel(
    const float* __restrict__ agg, const float* __restrict__ ab,
    const float* __restrict__ res, const float* __restrict__ g,
    const float* __restrict__ b, float* __restrict__ hout,
    const int* __restrict__ batch, float* __restrict__ gpool,
    float* __restrict__ cnt, int N_, int do_pool) {
  int n = blockIdx.x * 8 + (threadIdx.x >> 5);
  int lane = threadIdx.x & 31;
  if (n >= N_) return;
  int c0 = lane * 4;
  float v[4], sum = 0.f, sq = 0.f;
#pragma unroll
  for (int j = 0; j < 4; ++j) {
    int c = c0 + j;
    float t = agg[(size_t)n * D + c] + ab[c];
    t = (t > 0.f) ? t : (expf(t) - 1.f);  // ELU
    t += res[(size_t)n * D + c];
    v[j] = t; sum += t; sq += t * t;
  }
#pragma unroll
  for (int m = 1; m < 32; m <<= 1) {
    sum += __shfl_xor(sum, m, 32);
    sq  += __shfl_xor(sq, m, 32);
  }
  float mu = sum * (1.f / D);
  float rstd = rsqrtf(sq * (1.f / D) - mu * mu + LN_EPS);
  int bidx = do_pool ? batch[n] : 0;
#pragma unroll
  for (int j = 0; j < 4; ++j) {
    int c = c0 + j;
    float o = (v[j] - mu) * rstd * g[c] + b[c];
    if (do_pool) atomicAdd(&gpool[(size_t)bidx * D + c], o);
    else hout[(size_t)n * D + c] = o;
  }
  if (do_pool && lane == 0) atomicAdd(&cnt[bidx], 1.f);
}

// readout MLP: relu(mean @ Wd1 + bd1) @ Wd2 + bd2  (wave per graph)
__global__ __launch_bounds__(256) void readout_kernel(
    const float* __restrict__ gpool, const float* __restrict__ cnt,
    const float* __restrict__ Wd1, const float* __restrict__ bd1,
    const float* __restrict__ Wd2, const float* __restrict__ bd2,
    float* __restrict__ out, int G_) {
  int gi = blockIdx.x * 8 + (threadIdx.x >> 5);
  int lane = threadIdx.x & 31;
  if (gi >= G_) return;
  float inv = 1.f / fmaxf(cnt[gi], 1.f);
  float part = 0.f;
#pragma unroll
  for (int t = 0; t < 2; ++t) {
    int j = lane * 2 + t;
    float acc = bd1[j];
    for (int c = 0; c < D; ++c)
      acc = fmaf(gpool[(size_t)gi * D + c] * inv, Wd1[c * DEC_HID + j], acc);
    acc = fmaxf(acc, 0.f);
    part = fmaf(acc, Wd2[j], part);
  }
#pragma unroll
  for (int m = 1; m < 32; m <<= 1) part += __shfl_xor(part, m, 32);
  if (lane == 0) out[gi] = part + bd2[0];
}

// ---------------------------------------------------------------------------
extern "C" void kernel_launch(void* const* d_in, const int* in_sizes, int n_in,
                              void* d_out, int out_size, void* d_ws,
                              size_t ws_size, hipStream_t stream) {
  (void)n_in; (void)ws_size;
  const float* x = (const float*)d_in[0];
  const int* ei = (const int*)d_in[1];
  const float* eattr = (const float*)d_in[2];
  const int* batch = (const int*)d_in[3];
  const float* Wl1 = (const float*)d_in[4];  const float* bl1 = (const float*)d_in[5];
  const float* Wr1 = (const float*)d_in[6];  const float* br1 = (const float*)d_in[7];
  const float* We1 = (const float*)d_in[8];
  const float* att1 = (const float*)d_in[9]; const float* ab1 = (const float*)d_in[10];
  const float* lg1 = (const float*)d_in[11]; const float* lb1 = (const float*)d_in[12];
  const float* Wres = (const float*)d_in[13]; const float* bres = (const float*)d_in[14];
  const float* Wl2 = (const float*)d_in[15]; const float* bl2 = (const float*)d_in[16];
  const float* Wr2 = (const float*)d_in[17]; const float* br2 = (const float*)d_in[18];
  const float* We2 = (const float*)d_in[19];
  const float* att2 = (const float*)d_in[20]; const float* ab2 = (const float*)d_in[21];
  const float* lg2 = (const float*)d_in[22]; const float* lb2 = (const float*)d_in[23];
  const float* Wd1 = (const float*)d_in[24]; const float* bd1 = (const float*)d_in[25];
  const float* Wd2 = (const float*)d_in[26]; const float* bd2 = (const float*)d_in[27];

  const int N = in_sizes[0] / NODE_DIM;  // 100000
  const int E = in_sizes[1] / 2;         // 400000
  const int G = out_size;                // 4096 (OUT_D == 1)
  const int EA = E + N;

  // workspace layout (~300 MB of float32)
  float* ws = (float*)d_ws;
  size_t off = 0;
  float* deg = ws + off;   off += (size_t)N;
  float* lattr = ws + off; off += (size_t)N * EDGE_DIM;
  float* xl = ws + off;    off += (size_t)N * D;
  float* xr = ws + off;    off += (size_t)N * D;
  float* hres = ws + off;  off += (size_t)N * D;
  float* h1 = ws + off;    off += (size_t)N * D;
  float* agg = ws + off;   off += (size_t)N * D;     // zero-group start
  float* ssum = ws + off;  off += (size_t)N * HNUM;
  unsigned* mkey = (unsigned*)(ws + off); off += (size_t)N * HNUM;  // zero-group end
  float* logits = ws + off; off += (size_t)EA * HNUM;
  float* gpool = ws + off;  off += (size_t)G * D;
  float* cnt = ws + off;    off += (size_t)G;

  const int edgeBlocks = (EA + 7) / 8;
  const size_t zgrp = ((size_t)N * D + (size_t)N * HNUM * 2) * sizeof(float);

  // self-loop edge attributes (fill_value='mean')
  (void)hipMemsetAsync(deg, 0, (size_t)(N + N * EDGE_DIM) * sizeof(float), stream);
  deg_loopattr_kernel<<<(E * 16 + 255) / 256, 256, 0, stream>>>(ei, eattr, deg, lattr, E);
  loop_div_kernel<<<(N * EDGE_DIM + 255) / 256, 256, 0, stream>>>(lattr, deg, N);

  // ---- block 1: transforms (K=38) ----
  const int g38Blocks = (N + 63) / 64;
  gemm_k38_wmma_kernel<<<g38Blocks, 256, 0, stream>>>(x, Wl1, bl1, xl, N);
  gemm_k38_wmma_kernel<<<g38Blocks, 256, 0, stream>>>(x, Wr1, br1, xr, N);
  gemm_k38_wmma_kernel<<<g38Blocks, 256, 0, stream>>>(x, Wres, bres, hres, N);

  // mkey=0 is a valid "-inf" seed here: every node has a self-loop edge.
  (void)hipMemsetAsync(agg, 0, zgrp, stream);
  edge_logits_kernel<<<edgeBlocks, 256, 0, stream>>>(ei, eattr, lattr, xl, xr, We1, att1, logits, mkey, E, N);
  edge_exp_kernel<<<(EA * HNUM + 255) / 256, 256, 0, stream>>>(ei, logits, mkey, ssum, E, N);
  edge_agg_kernel<<<edgeBlocks, 256, 0, stream>>>(ei, logits, ssum, xl, agg, E, N);
  node_post_kernel<<<(N + 7) / 8, 256, 0, stream>>>(agg, ab1, hres, lg1, lb1, h1, batch, gpool, cnt, N, 0);

  // ---- block 2: transforms (K=128, async-LDS staged A) ----
  const int g128Blocks = (N + 31) / 32;
  gemm_k128_wmma_async_kernel<<<g128Blocks, 256, 0, stream>>>(h1, Wl2, bl2, xl, N);
  gemm_k128_wmma_async_kernel<<<g128Blocks, 256, 0, stream>>>(h1, Wr2, br2, xr, N);

  (void)hipMemsetAsync(agg, 0, zgrp, stream);
  (void)hipMemsetAsync(gpool, 0, (size_t)(G * D + G) * sizeof(float), stream);
  edge_logits_kernel<<<edgeBlocks, 256, 0, stream>>>(ei, eattr, lattr, xl, xr, We2, att2, logits, mkey, E, N);
  edge_exp_kernel<<<(EA * HNUM + 255) / 256, 256, 0, stream>>>(ei, logits, mkey, ssum, E, N);
  edge_agg_kernel<<<edgeBlocks, 256, 0, stream>>>(ei, logits, ssum, xl, agg, E, N);
  node_post_kernel<<<(N + 7) / 8, 256, 0, stream>>>(agg, ab2, h1, lg2, lb2, nullptr, batch, gpool, cnt, N, 1);

  readout_kernel<<<(G + 7) / 8, 256, 0, stream>>>(gpool, cnt, Wd1, bd1, Wd2, bd2, (float*)d_out, G);
}
